// Mamba2Layer_88304527606386
// MI455X (gfx1250) — matrix operand order repaired and checked by reading, not compiled, not run on hardware
//
#include <hip/hip_runtime.h>
#include <math.h>

// ---------------------------------------------------------------------------
// CDNA5 / gfx1250 implementation of the 3-timescale SSM + sparse-attention
// layer.  Big matmuls: v_wmma_f32_16x16x32_bf16 fed from LDS, staged by
// async-to-LDS copies (ASYNCcnt) + Tensor Data Mover (TENSORcnt), double
// buffered.
// ---------------------------------------------------------------------------

typedef __attribute__((ext_vector_type(16))) __bf16 v16bf;
typedef __attribute__((ext_vector_type(8)))  __bf16 v8bf;
typedef __attribute__((ext_vector_type(8)))  float  v8f;
typedef __attribute__((ext_vector_type(4)))  unsigned int u32x4;
typedef __attribute__((ext_vector_type(8)))  int i32x8;
typedef __attribute__((ext_vector_type(4)))  int i32x4;

#define LN_EPS 1e-5f
#define D_MODEL 1024
#define SEQ     2048
#define BATCH   4
#define NROW    (BATCH * SEQ)          // 8192
#define NEL     ((long long)NROW * D_MODEL)
#define KV_KEEP 614                     // max(1, int(2048 * 0.3))

// GEMM tiling
#define BM 128
#define BN 64
#define BK 32
#define LDS_ROW_BYTES 80                // 64B of data + 16B pad (bank spread)
#define LDS_ROW_ELEM  40
#define A_BUF_BYTES (BM * LDS_ROW_BYTES)            // 10240
#define B_BUF_BYTES (BN * LDS_ROW_BYTES)            // 5120
#define BUF_STRIDE  (A_BUF_BYTES + B_BUF_BYTES)     // 15360

#if __has_builtin(__builtin_amdgcn_tensor_load_to_lds)
#define USE_TDM_B 1
#else
#define USE_TDM_B 0
#endif

static __device__ __forceinline__ float sigmoidf_(float x) { return 1.0f / (1.0f + expf(-x)); }

static __device__ __forceinline__ void softmax3(const float* __restrict__ w, float o[3]) {
  float a = w[0], b = w[1], c = w[2];
  float m = fmaxf(a, fmaxf(b, c));
  float ea = expf(a - m), eb = expf(b - m), ec = expf(c - m);
  float s = ea + eb + ec;
  o[0] = ea / s; o[1] = eb / s; o[2] = ec / s;
}

// A-matrix fragment (16x32 bf16): per-lane two 16B chunks at +0 and +16 elems.
static __device__ __forceinline__ v16bf frag_interleaved(const __bf16* p) {
  v8bf lo = *(const v8bf*)(p);
  v8bf hi = *(const v8bf*)(p + 16);
  v16bf r;
#pragma unroll
  for (int i = 0; i < 8; ++i) { r[i] = lo[i]; r[i + 8] = hi[i]; }
  return r;
}

// B-matrix fragment: 16 contiguous K elements per lane (two 16B chunks).
static __device__ __forceinline__ v16bf frag_contig(const __bf16* p) {
  v8bf lo = *(const v8bf*)(p);
  v8bf hi = *(const v8bf*)(p + 8);
  v16bf r;
#pragma unroll
  for (int i = 0; i < 8; ++i) { r[i] = lo[i]; r[i + 8] = hi[i]; }
  return r;
}

// async global->LDS copy of one 16B chunk (tracked with ASYNCcnt)
static __device__ __forceinline__ void async_copy_b128(unsigned lds_addr, const void* gaddr) {
  asm volatile("global_load_async_to_lds_b128 %0, %1, off"
               :: "v"(lds_addr), "v"(gaddr) : "memory");
}
static __device__ __forceinline__ void wait_asynccnt0() {
  asm volatile("s_wait_asynccnt 0x0" ::: "memory");
}

// ---------------------------------------------------------------------------
// Batched GEMM:  C[b] = alpha * A[b] (MxK bf16) * B[b]^T (NxK bf16) (+bias)
// Block 256 thr = 8 waves; tile 128x64; wave tile 32x32 (2x2 WMMA 16x16x32).
// A tile staged via global_load_async_to_lds_b128; B tile via the Tensor
// Data Mover (tensor_load_to_lds) with DMA-inserted 16B row padding.
// Double-buffered LDS, one barrier per K step.
// ---------------------------------------------------------------------------
__global__ __launch_bounds__(256)
void gemm_bf16_kernel(const __bf16* __restrict__ A, const __bf16* __restrict__ B,
                      const float* __restrict__ bias, float* __restrict__ C,
                      int M, int N, int K,
                      long long sA, long long sB, long long sC, float alpha) {
  __shared__ __align__(16) char smem[2 * BUF_STRIDE];

  const int bz = blockIdx.z;
  A += (long long)bz * sA;
  B += (long long)bz * sB;
  C += (long long)bz * sC;

  const int tid  = threadIdx.x;
  const int lane = tid & 31;
  const int wid  = tid >> 5;
  const int wm   = wid & 3;                 // 4 waves along M
  const int wn   = wid >> 2;                // 2 waves along N
  const int blockRow = blockIdx.y * BM;
  const int blockCol = blockIdx.x * BN;
  const int rowBase = blockRow + wm * 32;
  const int colBase = blockCol + wn * 32;

  const unsigned smemBase = (unsigned)(unsigned long long)(&smem[0]); // LDS byte offset

  // --- tile staging -------------------------------------------------------
  auto issue_tile = [&](int buf, int k0) {
    const unsigned bufOff = smemBase + (unsigned)buf * BUF_STRIDE;
    // A tile: 128 rows x 64B = 512 x 16B chunks, 2 per thread (ASYNC path)
#pragma unroll
    for (int j = 0; j < 2; ++j) {
      const int c   = tid + j * 256;
      const int row = c >> 2;
      const int q   = c & 3;
      const char* g = (const char*)(A + (long long)(blockRow + row) * K + k0) + q * 16;
      async_copy_b128(bufOff + (unsigned)(row * LDS_ROW_BYTES + q * 16), g);
    }
#if USE_TDM_B
    // B tile: 64 x 32 bf16 via TDM; DMA pads each 64B row with 16B in LDS.
    if (wid == 0) {
      unsigned long long gaddr =
          (unsigned long long)(const char*)(B + (long long)blockCol * K + k0);
      u32x4 g0;
      g0[0] = 1u;                                            // count=1, user desc
      g0[1] = bufOff + A_BUF_BYTES;                          // lds_addr
      g0[2] = (unsigned)(gaddr & 0xFFFFFFFFu);               // global_addr[31:0]
      g0[3] = (unsigned)((gaddr >> 32) & 0x01FFFFFFu)        // global_addr[56:32]
              | (2u << 30);                                  // type=2 (image)
      i32x8 g1;
      g1[0] = (int)((1u << 16)        // data_size = 2 bytes
                  | (1u << 20)        // pad_enable
                  | (3u << 22)        // pad_interval: every 64B
                  | (3u << 25));      // pad_amount: 16B
      g1[1] = (int)(((unsigned)K & 0xFFFFu) << 16);          // tensor_dim0 lo16
      g1[2] = (int)(((unsigned)K >> 16) |
                    (((unsigned)N & 0xFFFFu) << 16));        // dim0 hi | dim1 lo
      g1[3] = (int)(((unsigned)N >> 16) | ((unsigned)BK << 16)); // dim1 hi | tile_dim0
      g1[4] = BN;                                            // tile_dim1 (tile_dim2=0)
      g1[5] = K;                                             // tensor_dim0_stride lo32
      g1[6] = 0;
      g1[7] = 0;
      i32x4 z; z[0] = 0; z[1] = 0; z[2] = 0; z[3] = 0;
      i32x8 z8; z8[0] = 0; z8[1] = 0; z8[2] = 0; z8[3] = 0;
      z8[4] = 0; z8[5] = 0; z8[6] = 0; z8[7] = 0;
      __builtin_amdgcn_tensor_load_to_lds(g0, g1, z, z, z8, 0);
    }
#else
    // Fallback: B tile via async copies, 256 x 16B chunks, 1 per thread.
    {
      const int row = tid >> 2;
      const int q   = tid & 3;
      const char* g = (const char*)(B + (long long)(blockCol + row) * K + k0) + q * 16;
      async_copy_b128(bufOff + A_BUF_BYTES + (unsigned)(row * LDS_ROW_BYTES + q * 16), g);
    }
#endif
  };

  // --- fragment geometry --------------------------------------------------
  const int lr   = lane & 15;
  const int hi16 = lane >> 4;
  const int kA   = hi16 * 8;                // A per-lane K offset
  const int kB   = hi16 * 8;                // B per-lane K offset

  v8f acc00 = 0.0f, acc01 = 0.0f, acc10 = 0.0f, acc11 = 0.0f;

  const int nsteps = K / BK;
  issue_tile(0, 0);

  for (int i = 0; i < nsteps; ++i) {
    const int cur = i & 1;
    wait_asynccnt0();
#if USE_TDM_B
    __builtin_amdgcn_s_wait_tensorcnt(0);
#endif
    __syncthreads();                         // publish buf 'cur'; retire reads of 1-cur
    if (i + 1 < nsteps) issue_tile(1 - cur, (i + 1) * BK);

    const __bf16* As = (const __bf16*)(smem + cur * BUF_STRIDE);
    const __bf16* Bs = (const __bf16*)(smem + cur * BUF_STRIDE + A_BUF_BYTES);

    v16bf a0 = frag_interleaved(As + (wm * 32 + lr)      * LDS_ROW_ELEM + kA);
    v16bf a1 = frag_interleaved(As + (wm * 32 + 16 + lr) * LDS_ROW_ELEM + kA);
    v16bf b0 = frag_interleaved(Bs + (wn * 32 + lr)      * LDS_ROW_ELEM + kB);
    v16bf b1 = frag_interleaved(Bs + (wn * 32 + 16 + lr) * LDS_ROW_ELEM + kB);

    acc00 = __builtin_amdgcn_wmma_f32_16x16x32_bf16(false, a0, false, b0, (short)0, acc00, false, false);
    acc01 = __builtin_amdgcn_wmma_f32_16x16x32_bf16(false, a0, false, b1, (short)0, acc01, false, false);
    acc10 = __builtin_amdgcn_wmma_f32_16x16x32_bf16(false, a1, false, b0, (short)0, acc10, false, false);
    acc11 = __builtin_amdgcn_wmma_f32_16x16x32_bf16(false, a1, false, b1, (short)0, acc11, false, false);
  }

  // C/D layout: VGPR j -> row (j + 8*hi16), col = lane & 15
  const int cn  = lane & 15;
  const int cm  = hi16 * 8;
  const int c0  = colBase + cn;
  const int c1  = colBase + 16 + cn;
  const float bb0 = bias ? bias[c0] : 0.0f;
  const float bb1 = bias ? bias[c1] : 0.0f;
#pragma unroll
  for (int j = 0; j < 8; ++j) {
    const int r0 = rowBase + cm + j;
    const int r1 = rowBase + 16 + cm + j;
    C[(long long)r0 * N + c0] = acc00[j] * alpha + bb0;
    C[(long long)r0 * N + c1] = acc01[j] * alpha + bb1;
    C[(long long)r1 * N + c0] = acc10[j] * alpha + bb0;
    C[(long long)r1 * N + c1] = acc11[j] * alpha + bb1;
  }
}

// ---------------------------------------------------------------------------
// Row LayerNorm (biased variance, eps 1e-5); optional bf16 copy of output.
// ---------------------------------------------------------------------------
__global__ __launch_bounds__(256)
void ln_kernel(const float* __restrict__ x, const float* __restrict__ w,
               const float* __restrict__ b, float* __restrict__ outF,
               __bf16* __restrict__ outBF, int d) {
  const long long row = blockIdx.x;
  const float* xr = x + row * d;
  __shared__ float red[256];
  const int tid = threadIdx.x;

  float s = 0.0f;
  for (int i = tid; i < d; i += 256) s += xr[i];
  red[tid] = s; __syncthreads();
  for (int off = 128; off > 0; off >>= 1) { if (tid < off) red[tid] += red[tid + off]; __syncthreads(); }
  const float mean = red[0] / (float)d;
  __syncthreads();

  float v = 0.0f;
  for (int i = tid; i < d; i += 256) { float t = xr[i] - mean; v += t * t; }
  red[tid] = v; __syncthreads();
  for (int off = 128; off > 0; off >>= 1) { if (tid < off) red[tid] += red[tid + off]; __syncthreads(); }
  const float rstd = rsqrtf(red[0] / (float)d + LN_EPS);
  __syncthreads();

  for (int i = tid; i < d; i += 256) {
    float val = (xr[i] - mean) * rstd * w[i] + b[i];
    outF[row * d + i] = val;
    if (outBF) outBF[row * d + i] = (__bf16)val;
  }
}

__global__ __launch_bounds__(256)
void cvt_bf16_kernel(const float* __restrict__ in, __bf16* __restrict__ out, long long n) {
  long long i = (long long)blockIdx.x * 256 + threadIdx.x;
  if (i < n) out[i] = (__bf16)in[i];
}

// ---------------------------------------------------------------------------
// gated = gate_scale * sum_g softmax(gw)[g] * sigmoid(glin_g + gate_b_g) * xn
//         + 0.1 * residual_scale * xn
// ---------------------------------------------------------------------------
__global__ __launch_bounds__(256)
void gate_combine_kernel(const float* __restrict__ glin3, const float* __restrict__ xn,
                         const float* __restrict__ gate_b, const float* __restrict__ gate_weights,
                         const float* __restrict__ gate_scale, const float* __restrict__ residual_scale,
                         float* __restrict__ gated, long long n, int d) {
  long long i = (long long)blockIdx.x * 256 + threadIdx.x;
  if (i >= n) return;
  float gw[3]; softmax3(gate_weights, gw);
  const int col = (int)(i % d);
  const float x = xn[i];
  float acc = 0.0f;
#pragma unroll
  for (int g = 0; g < 3; ++g) {
    float z = glin3[(long long)g * n + i] + gate_b[g * d + col];
    acc += gw[g] * sigmoidf_(z) * x;
  }
  gated[i] = acc * gate_scale[0] + x * residual_scale[0] * 0.1f;
}

// ---------------------------------------------------------------------------
// uB[row, j] = dot(xn_t[row, :], Bm[j, :])  for j in 0..15.
// ---------------------------------------------------------------------------
__global__ __launch_bounds__(256)
void ub_kernel(const float* __restrict__ xn, const float* __restrict__ Bm,
               float* __restrict__ uB, int d) {
  const long long row = blockIdx.x;
  const float* u = xn + row * d;
  __shared__ float red[256];
  const int tid = threadIdx.x;
  const int j = tid >> 4;
  const int l = tid & 15;
  float s = 0.0f;
  for (int i = l; i < d; i += 16) s += u[i] * Bm[j * d + i];
  red[tid] = s; __syncthreads();
  for (int off = 8; off > 0; off >>= 1) { if (l < off) red[tid] += red[tid + off]; __syncthreads(); }
  if (l == 0) uB[row * 16 + j] = red[tid];
}

// ---------------------------------------------------------------------------
// Sequential recurrence: h_t = tanh(h_{t-1} @ A + uB_t); one wave per batch.
// ---------------------------------------------------------------------------
__global__ __launch_bounds__(32)
void scan_kernel(const float* __restrict__ uB, const float* __restrict__ A,
                 float* __restrict__ H, int S) {
  const int b = blockIdx.x;
  const int j = threadIdx.x;
  const int jj = j & 15;
  float acol[16];
#pragma unroll
  for (int k = 0; k < 16; ++k) acol[k] = A[k * 16 + jj];   // column jj of A

  const float* ub = uB + (long long)b * S * 16;
  float* Hb = H + (long long)b * S * 16;
  float h = 0.0f;
  for (int s = 0; s < S; ++s) {
    float acc = ub[s * 16 + jj];
#pragma unroll
    for (int k = 0; k < 16; ++k) acc += acol[k] * __shfl(h, k, 32);
    h = tanhf(acc);
    if (j < 16) Hb[s * 16 + j] = h;
  }
}

// ---------------------------------------------------------------------------
// outAcc += tw[t]*ssm_scale * ( H@C + (D+1)*xn_t )   (init at t==0)
// ---------------------------------------------------------------------------
__global__ __launch_bounds__(256)
void ssm_combine_kernel(const float* __restrict__ H, const float* __restrict__ C,
                        const float* __restrict__ Dv, const float* __restrict__ xn_t,
                        const float* __restrict__ tw_all, const float* __restrict__ ssm_scale,
                        int t, float* __restrict__ outAcc, int d) {
  const long long row = blockIdx.x;
  __shared__ float hs[16];
  if (threadIdx.x < 16) hs[threadIdx.x] = H[row * 16 + threadIdx.x];
  __syncthreads();
  float tw[3]; softmax3(tw_all, tw);
  const float scale = tw[t] * ssm_scale[0];
  for (int c = threadIdx.x; c < d; c += 256) {
    float acc = 0.0f;
#pragma unroll
    for (int j = 0; j < 16; ++j) acc += hs[j] * C[j * d + c];
    float val = scale * (acc + (Dv[c] + 1.0f) * xn_t[row * d + c]);
    if (t == 0) outAcc[row * d + c] = val;
    else        outAcc[row * d + c] += val;
  }
}

// ---------------------------------------------------------------------------
// Split fused QKV (fp32 [row,3d]) into q,k bf16 row-major and v^T bf16 [b,d,S].
// ---------------------------------------------------------------------------
__global__ __launch_bounds__(256)
void qkv_split_kernel(const float* __restrict__ qkv, __bf16* __restrict__ q,
                      __bf16* __restrict__ k, __bf16* __restrict__ vT,
                      int S, int d, long long n) {
  long long i = (long long)blockIdx.x * 256 + threadIdx.x;
  if (i >= n) return;
  const long long row = i / d;
  const int c = (int)(i % d);
  const int b = (int)(row / S);
  const int s = (int)(row % S);
  const float* src = qkv + row * (3 * d);
  q[i] = (__bf16)src[c];
  k[i] = (__bf16)src[d + c];
  vT[((long long)b * d + c) * S + s] = (__bf16)src[2 * d + c];
}

// ---------------------------------------------------------------------------
// Per-row top-k(614) threshold select + softmax -> sparse bf16 prob row.
// ---------------------------------------------------------------------------
__global__ __launch_bounds__(256)
void topk_softmax_kernel(const float* __restrict__ scores, __bf16* __restrict__ P,
                         int S, int kv) {
  const long long row = blockIdx.x;
  const float* sr = scores + row * (long long)S;
  __shared__ float vals[SEQ];
  __shared__ float red[256];
  const int tid = threadIdx.x;

  float mx = -3.4e38f, mn = 3.4e38f;
  for (int i = tid; i < S; i += 256) {
    float v = sr[i];
    vals[i] = v;
    mx = fmaxf(mx, v);
    mn = fminf(mn, v);
  }
  red[tid] = mx; __syncthreads();
  for (int off = 128; off > 0; off >>= 1) { if (tid < off) red[tid] = fmaxf(red[tid], red[tid + off]); __syncthreads(); }
  const float rowMax = red[0]; __syncthreads();
  red[tid] = mn; __syncthreads();
  for (int off = 128; off > 0; off >>= 1) { if (tid < off) red[tid] = fminf(red[tid], red[tid + off]); __syncthreads(); }
  const float rowMin = red[0]; __syncthreads();

  float lo = rowMin, hi = rowMax;
  for (int it = 0; it < 30; ++it) {
    float mid = 0.5f * (lo + hi);
    float c = 0.0f;
    for (int i = tid; i < S; i += 256) c += (vals[i] >= mid) ? 1.0f : 0.0f;
    red[tid] = c; __syncthreads();
    for (int off = 128; off > 0; off >>= 1) { if (tid < off) red[tid] += red[tid + off]; __syncthreads(); }
    float cnt = red[0]; __syncthreads();
    if (cnt > (float)kv) lo = mid; else hi = mid;
  }
  const float thr = lo;

  float s = 0.0f;
  for (int i = tid; i < S; i += 256)
    if (vals[i] >= thr) s += expf(vals[i] - rowMax);
  red[tid] = s; __syncthreads();
  for (int off = 128; off > 0; off >>= 1) { if (tid < off) red[tid] += red[tid + off]; __syncthreads(); }
  const float inv = 1.0f / red[0]; __syncthreads();

  for (int i = tid; i < S; i += 256) {
    float p = (vals[i] >= thr) ? expf(vals[i] - rowMax) * inv : 0.0f;
    P[row * (long long)S + i] = (__bf16)p;
  }
}

__global__ __launch_bounds__(256)
void attn_mix_kernel(const float* __restrict__ attnproj, const float* __restrict__ xn_attn,
                     const float* __restrict__ memory_gate, const float* __restrict__ attention_scale,
                     const float* __restrict__ ssmc, float* __restrict__ memg, long long n) {
  long long i = (long long)blockIdx.x * 256 + threadIdx.x;
  if (i >= n) return;
  float mem = attnproj[i] + xn_attn[i];
  memg[i] = mem * sigmoidf_(memory_gate[0]) * attention_scale[0] + ssmc[i] * 0.5f;
}

__global__ __launch_bounds__(256)
void final_kernel(const float* __restrict__ proj, const float* __restrict__ xn_pre,
                  const float* __restrict__ residual_scale, const float* __restrict__ grad_scale,
                  float* __restrict__ out, long long n) {
  long long i = (long long)blockIdx.x * 256 + threadIdx.x;
  if (i >= n) return;
  out[i] = (proj[i] + xn_pre[i] * residual_scale[0]) * grad_scale[0];
}

// ---------------------------------------------------------------------------
// Host orchestration
// ---------------------------------------------------------------------------
extern "C" void kernel_launch(void* const* d_in, const int* in_sizes, int n_in,
                              void* d_out, int out_size, void* d_ws, size_t ws_size,
                              hipStream_t stream) {
  (void)in_sizes; (void)n_in; (void)out_size; (void)ws_size;

  const float* x_in        = (const float*)d_in[0];
  const float* pre_ln_w    = (const float*)d_in[1];
  const float* pre_ln_b    = (const float*)d_in[2];
  const float* gate_w      = (const float*)d_in[3];
  const float* gate_b      = (const float*)d_in[4];
  const float* gate_wts    = (const float*)d_in[5];
  const float* gate_scale  = (const float*)d_in[6];
  const float* resid_scale = (const float*)d_in[7];
  const float* tsw         = (const float*)d_in[8];
  const float* ssm_A       = (const float*)d_in[9];
  const float* ssm_B       = (const float*)d_in[10];
  const float* ssm_C       = (const float*)d_in[11];
  const float* ssm_D       = (const float*)d_in[12];
  const float* ssm_ln_w    = (const float*)d_in[13];
  const float* ssm_ln_b    = (const float*)d_in[14];
  const float* ssm_scale   = (const float*)d_in[15];
  const float* qkv_w       = (const float*)d_in[16];
  const float* qkv_b       = (const float*)d_in[17];
  const float* attn_out_w  = (const float*)d_in[18];
  const float* attn_out_b  = (const float*)d_in[19];
  const float* attn_ln_w   = (const float*)d_in[20];
  const float* attn_ln_b   = (const float*)d_in[21];
  const float* memory_gate = (const float*)d_in[22];
  const float* attn_scale  = (const float*)d_in[23];
  const float* post_ln_w   = (const float*)d_in[24];
  const float* post_ln_b   = (const float*)d_in[25];
  const float* out_w       = (const float*)d_in[26];
  const float* out_b       = (const float*)d_in[27];
  const float* grad_scale  = (const float*)d_in[28];

  const int d = D_MODEL, S = SEQ;
  const long long nel = NEL;                       // 8192 * 1024
  const long long SS  = (long long)BATCH * S * S;  // 16.8M

  char* cur = (char*)d_ws;
  auto alloc = [&](size_t bytes) -> void* {
    void* p = (void*)cur;
    cur += (bytes + 255) & ~(size_t)255;
    return p;
  };

  float*  XN_F     = (float*) alloc(nel * 4);
  __bf16* XN_BF    = (__bf16*)alloc(nel * 2);
  float*  GLIN3    = (float*) alloc(3 * nel * 4);
  float*  GATED    = (float*) alloc(nel * 4);
  float*  XNT      = (float*) alloc(nel * 4);
  float*  UB       = (float*) alloc((size_t)NROW * 16 * 4);
  float*  Hst      = (float*) alloc((size_t)NROW * 16 * 4);
  float*  SSMC     = (float*) alloc(nel * 4);
  float*  XNA_F    = (float*) alloc(nel * 4);
  __bf16* XNA_BF   = (__bf16*)alloc(nel * 2);
  float*  QKV      = (float*) alloc(3 * nel * 4);
  __bf16* QBF      = (__bf16*)alloc(nel * 2);
  __bf16* KBF      = (__bf16*)alloc(nel * 2);
  __bf16* VTBF     = (__bf16*)alloc(nel * 2);
  float*  SCORES   = (float*) alloc((size_t)SS * 4);
  __bf16* PBF      = (__bf16*)alloc((size_t)SS * 2);
  float*  ATTNV    = (float*) alloc(nel * 4);
  __bf16* ATTNV_BF = (__bf16*)alloc(nel * 2);
  float*  PROJ     = (float*) alloc(nel * 4);
  float*  MEMG     = (float*) alloc(nel * 4);
  float*  POST_F   = (float*) alloc(nel * 4);
  __bf16* POST_BF  = (__bf16*)alloc(nel * 2);
  __bf16* GATEW_BF = (__bf16*)alloc((size_t)3 * d * d * 2);
  __bf16* QKVW_BF  = (__bf16*)alloc((size_t)3 * d * d * 2);
  __bf16* AOW_BF   = (__bf16*)alloc((size_t)d * d * 2);
  __bf16* OUTW_BF  = (__bf16*)alloc((size_t)d * d * 2);

  const dim3 blk(256);
  auto eg = [](long long n) { return dim3((unsigned)((n + 255) / 256)); };

  // --- weight conversions to bf16 ---
  cvt_bf16_kernel<<<eg(3LL * d * d), blk, 0, stream>>>(gate_w,     GATEW_BF, 3LL * d * d);
  cvt_bf16_kernel<<<eg(3LL * d * d), blk, 0, stream>>>(qkv_w,      QKVW_BF,  3LL * d * d);
  cvt_bf16_kernel<<<eg((long long)d * d), blk, 0, stream>>>(attn_out_w, AOW_BF, (long long)d * d);
  cvt_bf16_kernel<<<eg((long long)d * d), blk, 0, stream>>>(out_w,      OUTW_BF, (long long)d * d);

  // --- 1. pre-LN ---
  ln_kernel<<<NROW, blk, 0, stream>>>(x_in, pre_ln_w, pre_ln_b, XN_F, XN_BF, d);

  // --- 2. gate projections (batched x3 WMMA GEMM) + combine ---
  gemm_bf16_kernel<<<dim3(d / BN, NROW / BM, 3), blk, 0, stream>>>(
      XN_BF, GATEW_BF, nullptr, GLIN3, NROW, d, d,
      0LL, (long long)d * d, nel, 1.0f);
  gate_combine_kernel<<<eg(nel), blk, 0, stream>>>(
      GLIN3, XN_F, gate_b, gate_wts, gate_scale, resid_scale, GATED, nel, d);

  // --- 3. three SSM timescales ---
  for (int t = 0; t < 3; ++t) {
    ln_kernel<<<NROW, blk, 0, stream>>>(GATED, ssm_ln_w + t * d, ssm_ln_b + t * d, XNT, nullptr, d);
    ub_kernel<<<NROW, blk, 0, stream>>>(XNT, ssm_B + (long long)t * 16 * d, UB, d);
    scan_kernel<<<BATCH, dim3(32), 0, stream>>>(UB, ssm_A + t * 256, Hst, S);
    ssm_combine_kernel<<<NROW, blk, 0, stream>>>(
        Hst, ssm_C + (long long)t * 16 * d, ssm_D + t * d, XNT, tsw, ssm_scale, t, SSMC, d);
  }

  // --- 4. sparse attention ---
  ln_kernel<<<NROW, blk, 0, stream>>>(SSMC, attn_ln_w, attn_ln_b, XNA_F, XNA_BF, d);
  gemm_bf16_kernel<<<dim3(3 * d / BN, NROW / BM, 1), blk, 0, stream>>>(
      XNA_BF, QKVW_BF, qkv_b, QKV, NROW, 3 * d, d, 0LL, 0LL, 0LL, 1.0f);
  qkv_split_kernel<<<eg(nel), blk, 0, stream>>>(QKV, QBF, KBF, VTBF, S, d, nel);

  // scores = q @ k^T / sqrt(d)   (batched over B)
  gemm_bf16_kernel<<<dim3(S / BN, S / BM, BATCH), blk, 0, stream>>>(
      QBF, KBF, nullptr, SCORES, S, S, d,
      (long long)S * d, (long long)S * d, (long long)S * S, 0.03125f);

  topk_softmax_kernel<<<NROW, blk, 0, stream>>>(SCORES, PBF, S, KV_KEEP);

  // attn = P @ v  == P (SxS) * vT^T (dxS)
  gemm_bf16_kernel<<<dim3(d / BN, S / BM, BATCH), blk, 0, stream>>>(
      PBF, VTBF, nullptr, ATTNV, S, d, S,
      (long long)S * S, (long long)d * S, (long long)S * d, 1.0f);

  cvt_bf16_kernel<<<eg(nel), blk, 0, stream>>>(ATTNV, ATTNV_BF, nel);
  gemm_bf16_kernel<<<dim3(d / BN, NROW / BM, 1), blk, 0, stream>>>(
      ATTNV_BF, AOW_BF, attn_out_b, PROJ, NROW, d, d, 0LL, 0LL, 0LL, 1.0f);

  attn_mix_kernel<<<eg(nel), blk, 0, stream>>>(
      PROJ, XNA_F, memory_gate, attn_scale, SSMC, MEMG, nel);

  // --- 5. output projection + residual ---
  ln_kernel<<<NROW, blk, 0, stream>>>(MEMG, post_ln_w, post_ln_b, POST_F, POST_BF, d);
  gemm_bf16_kernel<<<dim3(d / BN, NROW / BM, 1), blk, 0, stream>>>(
      POST_BF, OUTW_BF, out_b, PROJ, NROW, d, d, 0LL, 0LL, 0LL, 1.0f);
  final_kernel<<<eg(nel), blk, 0, stream>>>(
      PROJ, XN_F, resid_scale, grad_scale, (float*)d_out, nel);
}
